// Attention_57543971831928
// MI455X (gfx1250) — matrix operand order repaired and checked by reading, not compiled, not run on hardware
//
#include <hip/hip_runtime.h>
#include <hip/hip_bf16.h>
#include <cstdint>

typedef __attribute__((ext_vector_type(16))) _Float16 v16h;
typedef __attribute__((ext_vector_type(8)))  float    v8f;
typedef __attribute__((ext_vector_type(4)))  int      v4i;

#define LN_EPS 1e-5f

// ---------------------------------------------------------------------------
// gfx1250 async global->LDS (ASYNCcnt-tracked copy), with fallback
// ---------------------------------------------------------------------------
#if __has_builtin(__builtin_amdgcn_global_load_async_to_lds_b128) && \
    __has_builtin(__builtin_amdgcn_s_wait_asynccnt)
#define USE_ASYNC_LDS 1
typedef __attribute__((address_space(1))) v4i gas_v4i;
typedef __attribute__((address_space(3))) v4i las_v4i;
#endif

__device__ __forceinline__ void cp_async16(_Float16* lds_dst, const _Float16* gsrc) {
#ifdef USE_ASYNC_LDS
  __builtin_amdgcn_global_load_async_to_lds_b128(
      (gas_v4i*)(uintptr_t)gsrc,
      (las_v4i*)(uintptr_t)lds_dst,
      0, 0);
#else
  *(uint4*)lds_dst = *(const uint4*)gsrc;
#endif
}

__device__ __forceinline__ void cp_async_wait() {
#ifdef USE_ASYNC_LDS
  __builtin_amdgcn_s_wait_asynccnt(0);
#endif
}

// ---------------------------------------------------------------------------
// WMMA fragment loaders (v_wmma_f32_16x16x32_f16 operand layouts, wave32)
// A 16x32 f16: lane(0-15)=M; K: VGPR0-3 -> half*8+{0..7}, VGPR4-7 -> 16+half*8+{0..7}
// B 32x16 f16: lane(0-15)=N; K: VGPR0-3 -> half*8+{0..7}, VGPR4-7 -> 16+half*8+{0..7}
// C/D 16x16 f32: lane(0-15)=N, VGPR i -> M = i + 8*half
// ---------------------------------------------------------------------------
__device__ __forceinline__ v16h load_frag_a(const _Float16* base, int pitch_h,
                                            int row0, int col0) {
  int lane = threadIdx.x & 31;
  int m = lane & 15;
  int hf = lane >> 4;
  const uint32_t* p = (const uint32_t*)base;
  int pr = pitch_h >> 1;
  int r = row0 + m;
  int cA = (col0 + hf * 8) >> 1;
  int cB = (col0 + 16 + hf * 8) >> 1;
  union { v16h h; uint32_t u[8]; } f;
#pragma unroll
  for (int g = 0; g < 4; ++g) {
    f.u[g]     = p[r * pr + cA + g];
    f.u[4 + g] = p[r * pr + cB + g];
  }
  return f.h;
}

__device__ __forceinline__ v16h load_frag_b(const _Float16* base, int pitch_h,
                                            int row0, int col0) {
  int lane = threadIdx.x & 31;
  int n = lane & 15;
  int hf = lane >> 4;
  const uint32_t* p = (const uint32_t*)base;
  int pr = pitch_h >> 1;
  int r = row0 + n;
  int cA = (col0 + hf * 8) >> 1;
  int cB = (col0 + 16 + hf * 8) >> 1;
  union { v16h h; uint32_t u[8]; } f;
#pragma unroll
  for (int g = 0; g < 4; ++g) {
    f.u[g]     = p[r * pr + cA + g];
    f.u[4 + g] = p[r * pr + cB + g];
  }
  return f.h;
}

// ---------------------------------------------------------------------------
// Kernel 1: weights f32 -> f16
// ---------------------------------------------------------------------------
__global__ void convert_w_kernel(const float* __restrict__ W1,
                                 const float* __restrict__ W2,
                                 _Float16* __restrict__ w1h,
                                 _Float16* __restrict__ w2h) {
  int i = blockIdx.x * 256 + threadIdx.x;
  const int n1 = 1152 * 384, n2 = 384 * 384;
  if (i < n1) w1h[i] = (_Float16)W1[i];
  else if (i < n1 + n2) w2h[i - n1] = (_Float16)W2[i - n1];
}

// ---------------------------------------------------------------------------
// Kernel 2: LayerNorm stats per token (coalesced over tokens)
// ---------------------------------------------------------------------------
__global__ __launch_bounds__(256)
void ln_stats_kernel(const float* __restrict__ x,
                     float* __restrict__ mu, float* __restrict__ rstd) {
  int t = blockIdx.x * 256 + threadIdx.x;   // 0..18431
  int b = t / 2304, n = t % 2304;
  const float* px = x + (size_t)b * 384 * 2304 + n;
  float s = 0.f, s2 = 0.f;
  for (int c = 0; c < 384; ++c) {
    float v = px[c * 2304];
    s += v; s2 += v * v;
  }
  float m = s * (1.0f / 384.0f);
  float var = s2 * (1.0f / 384.0f) - m * m;
  mu[t] = m;
  rstd[t] = rsqrtf(var + LN_EPS);
}

// ---------------------------------------------------------------------------
// Kernel 3: normalize + transpose [B,C,N] -> tnorm f16 [B,N,C] via LDS tile
// ---------------------------------------------------------------------------
__global__ __launch_bounds__(256)
void ln_norm_kernel(const float* __restrict__ x, const float* __restrict__ mu,
                    const float* __restrict__ rstd, _Float16* __restrict__ tnorm) {
  __shared__ float tile[32][33];
  int blk = blockIdx.x;
  int ct = blk % 12;
  int nt = (blk / 12) % 72;
  int b  = blk / (12 * 72);
  int tid = threadIdx.x;
  int nn = tid & 31, cc = tid >> 5;
#pragma unroll
  for (int s = 0; s < 4; ++s) {
    int c = ct * 32 + cc + s * 8;
    tile[cc + s * 8][nn] = x[((size_t)b * 384 + c) * 2304 + nt * 32 + nn];
  }
  __syncthreads();
  int tc = tid & 31, rr = tid >> 5;
#pragma unroll
  for (int s = 0; s < 4; ++s) {
    int nl = rr + s * 8;
    int n = nt * 32 + nl;
    float v = (tile[tc][nl] - mu[b * 2304 + n]) * rstd[b * 2304 + n];
    tnorm[((size_t)b * 2304 + n) * 384 + ct * 32 + tc] = (_Float16)v;
  }
}

// ---------------------------------------------------------------------------
// Kernel 4/6: WMMA GEMM  out[M,Nd] = A[M,384] @ W[Nd,384]^T + bias (+epilogue)
// BM=128 x BN=64 tile per workgroup (4 waves); each wave: 32 rows x 64 cols.
// BK=64, double-buffered async staging: tile s+1 loads overlap tile s WMMAs.
// mode 0: f16 store (QKV);  mode 1: f32 store + LN residual, transposed layout
// ---------------------------------------------------------------------------
__global__ __launch_bounds__(128)
void gemm_ep_kernel(const _Float16* __restrict__ A, const _Float16* __restrict__ Wt,
                    int Nd, const float* __restrict__ bias, int mode,
                    _Float16* __restrict__ out_h, float* __restrict__ out_f,
                    const float* __restrict__ x, const float* __restrict__ mu,
                    const float* __restrict__ rstd) {
  const int K = 384;
  const int PIT = 72;                      // LDS row pitch in halves (144B, 16B-aligned)
  __shared__ _Float16 Asub[2][128 * PIT];
  __shared__ _Float16 Bsub[2][64 * PIT];
  const int Mtiles = 18432 / 128;
  int blk = blockIdx.x;
  int m0 = (blk % Mtiles) * 128;
  int n0 = (blk / Mtiles) * 64;
  int tid = threadIdx.x;
  int wave = tid >> 5;
  v8f acc[2][4] = {};

  auto stage = [&](int ks, int buf) {
#pragma unroll
    for (int it = 0; it < 8; ++it) {       // A: 128 rows x 64 halves, 8x16B per thread
      int idx = it * 128 + tid;
      int r = idx >> 3, c = idx & 7;
      cp_async16(&Asub[buf][r * PIT + c * 8], A + (size_t)(m0 + r) * K + ks + c * 8);
    }
#pragma unroll
    for (int it = 0; it < 4; ++it) {       // B: 64 rows x 64 halves, 4x16B per thread
      int idx = it * 128 + tid;
      int r = idx >> 3, c = idx & 7;
      cp_async16(&Bsub[buf][r * PIT + c * 8], Wt + (size_t)(n0 + r) * K + ks + c * 8);
    }
  };

  stage(0, 0);
  cp_async_wait();
  __syncthreads();

  for (int s = 0; s < 6; ++s) {
    int cur = s & 1, nxt = cur ^ 1;
    bool hn = (s + 1 < 6);
    if (hn) stage((s + 1) * 64, nxt);      // overlap next-stage loads with compute

#pragma unroll
    for (int kk = 0; kk < 64; kk += 32) {
      v16h af0 = load_frag_a(Asub[cur], PIT, wave * 32, kk);
      v16h af1 = load_frag_a(Asub[cur], PIT, wave * 32 + 16, kk);
#pragma unroll
      for (int t = 0; t < 4; ++t) {
        v16h bf = load_frag_b(Bsub[cur], PIT, t * 16, kk);
        acc[0][t] = __builtin_amdgcn_wmma_f32_16x16x32_f16(false, af0, false, bf,
                                                           (short)0, acc[0][t], false, false);
        acc[1][t] = __builtin_amdgcn_wmma_f32_16x16x32_f16(false, af1, false, bf,
                                                           (short)0, acc[1][t], false, false);
      }
    }

    if (hn) {
      cp_async_wait();
      __syncthreads();
    }
  }

  int lane = tid & 31;
  int hf = lane >> 4, nl = lane & 15;
#pragma unroll
  for (int t = 0; t < 4; ++t) {
    int c = n0 + t * 16 + nl;
    float bv = bias[c];
#pragma unroll
    for (int wr = 0; wr < 2; ++wr) {
#pragma unroll
      for (int i = 0; i < 8; ++i) {
        int r = m0 + wave * 32 + wr * 16 + hf * 8 + i;
        float v = acc[wr][t][i] + bv;
        if (mode == 0) {
          out_h[(size_t)r * Nd + c] = (_Float16)v;
        } else {
          int b = r / 2304, tok = r % 2304;
          float tn = (x[((size_t)b * 384 + c) * 2304 + tok] - mu[b * 2304 + tok]) *
                     rstd[b * 2304 + tok];
          out_f[((size_t)b * 384 + c) * 2304 + tok] = v + tn;
        }
      }
    }
  }
}

// ---------------------------------------------------------------------------
// Kernel 5: fused flash attention, per (b, head, 64-row Q block)
// Double-buffered K (async-to-LDS) and V (vector load + transposed scatter):
// tile j+1 memory ops overlap tile j WMMAs + softmax.
// ---------------------------------------------------------------------------
__global__ __launch_bounds__(128)
void flash_attn_kernel(const _Float16* __restrict__ qkv, _Float16* __restrict__ obuf) {
  const int NT = 2304, P3 = 1152, PIT = 72;
  __shared__ _Float16 Kt[2][64 * PIT];     // [token][d]
  __shared__ _Float16 Vt[2][64 * PIT];     // [d][token] (transposed)
  __shared__ _Float16 Pb[4 * 16 * PIT];    // per-wave P scratch
  int blk = blockIdx.x;
  int q0 = (blk % 36) * 64;
  int bh = blk / 36;
  int b = bh / 6, h = bh % 6;
  int tid = threadIdx.x;
  int wave = tid >> 5, lane = tid & 31;
  int hf = lane >> 4, nl = lane & 15;

  const _Float16* qbase = qkv + (size_t)(b * NT + q0) * P3 + h * 64;
  v16h qf0 = load_frag_a(qbase, P3, wave * 16, 0);
  v16h qf1 = load_frag_a(qbase, P3, wave * 16, 32);

  v8f accO[4] = {};
  float mrun[8], lrun[8];
#pragma unroll
  for (int i = 0; i < 8; ++i) { mrun[i] = -3.0e38f; lrun[i] = 0.f; }
  _Float16* Pw = Pb + wave * 16 * PIT;
  const float sc = 0.125f;   // 1/sqrt(64)

  auto stageK = [&](int j0, int buf) {
#pragma unroll
    for (int it = 0; it < 4; ++it) {
      int idx = it * 128 + tid;
      int tok = idx >> 3, c = idx & 7;
      cp_async16(&Kt[buf][tok * PIT + c * 8],
                 qkv + (size_t)(b * NT + j0 + tok) * P3 + 384 + h * 64 + c * 8);
    }
  };
  auto loadV = [&](int j0, uint4* vv) {
#pragma unroll
    for (int it = 0; it < 4; ++it) {
      int idx = it * 128 + tid;
      int tok = idx >> 3, c = idx & 7;
      vv[it] = *(const uint4*)(qkv + (size_t)(b * NT + j0 + tok) * P3 + 768 +
                               h * 64 + c * 8);
    }
  };
  auto scatterV = [&](const uint4* vv, int buf) {
#pragma unroll
    for (int it = 0; it < 4; ++it) {
      int idx = it * 128 + tid;
      int tok = idx >> 3, c = idx & 7;
      const _Float16* hs = (const _Float16*)&vv[it];
#pragma unroll
      for (int j = 0; j < 8; ++j)
        Vt[buf][(c * 8 + j) * PIT + tok] = hs[j];
    }
  };

  // prologue: stage tile 0
  {
    stageK(0, 0);
    uint4 v0[4];
    loadV(0, v0);
    scatterV(v0, 0);
    cp_async_wait();
    __syncthreads();
  }

  for (int s = 0; s < 36; ++s) {
    int j0 = s * 64;
    int cur = s & 1, nxt = cur ^ 1;
    bool hn = (s + 1 < 36);
    uint4 vnext[4];
    if (hn) {                       // overlap next tile's memory ops with compute
      stageK(j0 + 64, nxt);
      loadV(j0 + 64, vnext);
    }

    // S = Q K^T
    v8f accS[4] = {};
#pragma unroll
    for (int t = 0; t < 4; ++t) {
      v16h kb0 = load_frag_b(Kt[cur], PIT, t * 16, 0);
      accS[t] = __builtin_amdgcn_wmma_f32_16x16x32_f16(false, qf0, false, kb0,
                                                       (short)0, accS[t], false, false);
      v16h kb1 = load_frag_b(Kt[cur], PIT, t * 16, 32);
      accS[t] = __builtin_amdgcn_wmma_f32_16x16x32_f16(false, qf1, false, kb1,
                                                       (short)0, accS[t], false, false);
    }

    // online softmax (row r = hf*8+i lives across lanes 0..15 of its half-wave)
    float mloc[8], rsum[8];
#pragma unroll
    for (int i = 0; i < 8; ++i) {
      float mv = fmaxf(fmaxf(accS[0][i], accS[1][i]), fmaxf(accS[2][i], accS[3][i]));
      mloc[i] = mv * sc;
    }
#pragma unroll
    for (int i = 0; i < 8; ++i) {
#pragma unroll
      for (int ss = 1; ss < 16; ss <<= 1)
        mloc[i] = fmaxf(mloc[i], __shfl_xor(mloc[i], ss, 32));
    }
#pragma unroll
    for (int i = 0; i < 8; ++i) {
      float mnew = fmaxf(mrun[i], mloc[i]);
      float corr = __expf(mrun[i] - mnew);
      lrun[i] *= corr;
#pragma unroll
      for (int t = 0; t < 4; ++t) accO[t][i] *= corr;
      float rs = 0.f;
#pragma unroll
      for (int t = 0; t < 4; ++t) {
        float pv = __expf(accS[t][i] * sc - mnew);
        rs += pv;
        Pw[(hf * 8 + i) * PIT + t * 16 + nl] = (_Float16)pv;
      }
      rsum[i] = rs;
      mrun[i] = mnew;
    }
#pragma unroll
    for (int i = 0; i < 8; ++i) {
#pragma unroll
      for (int ss = 1; ss < 16; ss <<= 1) rsum[i] += __shfl_xor(rsum[i], ss, 32);
      lrun[i] += rsum[i];
    }

    __builtin_amdgcn_wave_barrier();  // per-wave LDS RAW; DS ops in-order per wave

    // O += P V
    v16h pf0 = load_frag_a(Pw, PIT, 0, 0);
    v16h pf1 = load_frag_a(Pw, PIT, 0, 32);
#pragma unroll
    for (int t = 0; t < 4; ++t) {
      v16h vb0 = load_frag_b(Vt[cur], PIT, t * 16, 0);
      accO[t] = __builtin_amdgcn_wmma_f32_16x16x32_f16(false, pf0, false, vb0,
                                                       (short)0, accO[t], false, false);
      v16h vb1 = load_frag_b(Vt[cur], PIT, t * 16, 32);
      accO[t] = __builtin_amdgcn_wmma_f32_16x16x32_f16(false, pf1, false, vb1,
                                                       (short)0, accO[t], false, false);
    }

    if (hn) {
      scatterV(vnext, nxt);  // land V tile j+1 before the barrier
      cp_async_wait();       // K tile j+1 async writes complete
      __syncthreads();       // publish tile j+1; also fences reads of buffers
    }
  }

#pragma unroll
  for (int i = 0; i < 8; ++i) {
    float inv = 1.0f / lrun[i];
    int r = b * NT + q0 + wave * 16 + hf * 8 + i;
#pragma unroll
    for (int t = 0; t < 4; ++t)
      obuf[(size_t)r * 384 + h * 64 + t * 16 + nl] = (_Float16)(accO[t][i] * inv);
  }
}

// ---------------------------------------------------------------------------
extern "C" void kernel_launch(void* const* d_in, const int* in_sizes, int n_in,
                              void* d_out, int out_size, void* d_ws, size_t ws_size,
                              hipStream_t stream) {
  (void)in_sizes; (void)n_in; (void)out_size; (void)ws_size;
  const float* x  = (const float*)d_in[0];
  const float* W1 = (const float*)d_in[1];
  const float* b1 = (const float*)d_in[2];
  const float* W2 = (const float*)d_in[3];
  const float* b2 = (const float*)d_in[4];
  float* out = (float*)d_out;

  char* ws = (char*)d_ws;
  size_t off = 0;
  auto alloc = [&](size_t bytes) -> void* {
    off = (off + 255) & ~(size_t)255;
    void* p = ws + off;
    off += bytes;
    return p;
  };
  _Float16* tnorm = (_Float16*)alloc(18432ull * 384 * 2);
  _Float16* qkv   = (_Float16*)alloc(18432ull * 1152 * 2);
  _Float16* obuf  = (_Float16*)alloc(18432ull * 384 * 2);
  _Float16* w1h   = (_Float16*)alloc(1152ull * 384 * 2);
  _Float16* w2h   = (_Float16*)alloc(384ull * 384 * 2);
  float* mu   = (float*)alloc(18432ull * 4);
  float* rstd = (float*)alloc(18432ull * 4);

  convert_w_kernel<<<2304, 256, 0, stream>>>(W1, W2, w1h, w2h);
  ln_stats_kernel<<<72, 256, 0, stream>>>(x, mu, rstd);
  ln_norm_kernel<<<6912, 256, 0, stream>>>(x, mu, rstd, tnorm);
  // QKV projection: M=18432 (144 tiles of 128), Nd=1152 (18 tiles of 64)
  gemm_ep_kernel<<<144 * 18, 128, 0, stream>>>(tnorm, w1h, 1152, b1, 0,
                                               qkv, nullptr, nullptr, nullptr, nullptr);
  // fused flash attention
  flash_attn_kernel<<<1728, 128, 0, stream>>>(qkv, obuf);
  // output projection + bias + LN residual + transpose back: Nd=384 (6 tiles)
  gemm_ep_kernel<<<144 * 6, 128, 0, stream>>>(obuf, w2h, 384, b2, 1,
                                              nullptr, out, x, mu, rstd);
}
